// GPT2Attention_30648886624929
// MI455X (gfx1250) — compile-verified, matrix operand order checked by reading, not run on hardware
//
#include <hip/hip_runtime.h>
#include <hip/hip_bf16.h>

// ---------------- types / helpers ----------------
typedef __attribute__((ext_vector_type(16))) __bf16 v16bf;
typedef __attribute__((ext_vector_type(8)))  float  v8f;

union Frag {
    v16bf v;
    unsigned short u[16];
    uint4 q[2];
};

#define B_ 2
#define S_ 2048
#define E_ 1024
#define H_ 16
#define D_ 64
#define NEG_BIG (-3.0e38f)

// LDS W-tile: 64 rows x 32 bf16, row stride padded to 40 elements (80B) so the
// 16 rows a ds_load_b128 wavefront touches hit distinct banks. Two buffers.
#define WROWS   64
#define WSTRIDE 40
#define WTILE   (WROWS * WSTRIDE)            // elements per buffer

__device__ __forceinline__ unsigned short f2bf(float f) {
    unsigned int u = __float_as_uint(f);
    u += 0x7FFFu + ((u >> 16) & 1u);           // round-to-nearest-even
    return (unsigned short)(u >> 16);
}

// Load a 16x32 bf16 WMMA fragment row (A layout; B uses the same per-lane
// pattern with N playing the role of M).  p points at
// base + row*ld + k0 + 8*half; this lane owns two contiguous 16B chunks.
__device__ __forceinline__ v16bf ldfrag(const unsigned short* p) {
    Frag f;
    f.q[0] = *(const uint4*)(p);
    f.q[1] = *(const uint4*)(p + 16);
    return f.v;
}

__device__ __forceinline__ v8f wmma_bf16(v16bf a, v16bf b, v8f c) {
    return __builtin_amdgcn_wmma_f32_16x16x32_bf16(
        false, a, false, b, (short)0, c, false, false);
}

// Async copy 16B global -> LDS (per lane), tracked by ASYNCcnt.
// lds_off = wave-relative LDS byte address (low 32 bits of generic pointer).
__device__ __forceinline__ void async_ld_b128(unsigned lds_off, const void* g) {
    asm volatile("global_load_async_to_lds_b128 %0, %1, off"
                 :: "v"(lds_off), "v"(g)
                 : "memory");
}
__device__ __forceinline__ void wait_async0() {
    asm volatile("s_wait_asynccnt 0" ::: "memory");
}

// ---------------- fp32 -> bf16 convert ----------------
__global__ void cvt_bf16_kernel(const float* __restrict__ src,
                                unsigned short* __restrict__ dst, int n) {
    int i = blockIdx.x * blockDim.x + threadIdx.x;
    if (i < n) dst[i] = f2bf(src[i]);
}

// ---------------- shared GEMM core ----------------
// Block = 256 threads = 8 waves; macro-tile 128(M) x 64(N); K = E_ = 1024.
// W tile (64x32) async-staged into double-buffered LDS, shared by all waves.
// Wave w computes rows m0 = mb*128 + w*16, accumulating 4 16x16 tiles.
__device__ __forceinline__ void gemm_core(const unsigned short* __restrict__ Abf,
                                          const unsigned short* __restrict__ Wbf,
                                          int m0, int n0,
                                          unsigned short* lds,   // 2*WTILE elems
                                          v8f acc[4]) {
    const int tid  = threadIdx.x;
    const int lane = tid & 31;
    const int half = lane >> 4;
    const int idx  = lane & 15;

    // async-copy mapping: 256 chunks of 8 elements cover the 64x32 tile
    const int crow   = tid >> 2;             // 0..63
    const int cpiece = tid & 3;              // 0..3 (8 elems each)
    const unsigned ldsWoff =
        (unsigned)(crow * WSTRIDE + cpiece * 8) * 2u;  // byte offset in buffer
    const unsigned ldsBase = (unsigned)(uintptr_t)lds; // LDS byte base
    const unsigned short* gW =
        Wbf + (size_t)(n0 + crow) * E_ + cpiece * 8;

    const unsigned short* arow = Abf + (size_t)(m0 + idx) * E_ + 8 * half;

    // prologue: stage k0 = 0
    async_ld_b128(ldsBase + ldsWoff, gW);

    for (int k0 = 0, buf = 0; k0 < E_; k0 += 32, buf ^= 1) {
        wait_async0();
        __syncthreads();                     // everyone's tile chunk visible
        if (k0 + 32 < E_)                    // prefetch next into other buffer
            async_ld_b128(ldsBase + (unsigned)((buf ^ 1) * WTILE * 2) + ldsWoff,
                          gW + k0 + 32);

        v16bf a = ldfrag(arow + k0);
        const unsigned short* wb = lds + buf * WTILE + 8 * half;
#pragma unroll
        for (int t = 0; t < 4; ++t) {
            v16bf b = ldfrag(wb + (t * 16 + idx) * WSTRIDE);
            acc[t] = wmma_bf16(a, b, acc[t]);
        }
        __syncthreads();                     // reads done before buffer reuse
    }
}

// ---------------- QKV projection GEMM ----------------
// qkv[m,f] = sum_e x[m,e]*W_attn[f,e] + b_attn[f];  M=4096, N=3072, K=1024.
// Epilogue scatters into Q[B,H,S,D] (scaled), K[B,H,S,D], Vt[B,H,D,S].
__global__ __launch_bounds__(256)
void qkv_gemm_kernel(const unsigned short* __restrict__ xbf,
                     const unsigned short* __restrict__ wbf,
                     const float* __restrict__ bias,
                     unsigned short* __restrict__ Q,
                     unsigned short* __restrict__ K,
                     unsigned short* __restrict__ Vt) {
    __shared__ unsigned short Wl[2 * WTILE];
    const int lane = threadIdx.x & 31;
    const int wave = threadIdx.x >> 5;
    const int half = lane >> 4;
    const int idx  = lane & 15;
    const int NG   = (3 * E_) / 64;          // 48
    const int ng   = blockIdx.x % NG;
    const int mb   = blockIdx.x / NG;        // 0..31
    const int m0   = mb * 128 + wave * 16;
    const int n0   = ng * 64;

    v8f acc[4] = {};
    gemm_core(xbf, wbf, m0, n0, Wl, acc);

#pragma unroll
    for (int t = 0; t < 4; ++t) {
        const int f  = n0 + t * 16 + idx;
        const float bv = bias[f];
#pragma unroll
        for (int r = 0; r < 8; ++r) {
            const int m   = r + 8 * half;
            const int row = m0 + m;                 // = b*S + s
            const int bb  = row >> 11;              // /S
            const int ss  = row & (S_ - 1);
            const float val = acc[t][r] + bv;
            if (f < E_) {
                const int h = f >> 6, d = f & 63;
                Q[(((size_t)(bb * H_ + h)) * S_ + ss) * D_ + d] =
                    f2bf(val * 0.125f);             // fold SCALE = D^-0.5
            } else if (f < 2 * E_) {
                const int f2 = f - E_;
                const int h = f2 >> 6, d = f2 & 63;
                K[(((size_t)(bb * H_ + h)) * S_ + ss) * D_ + d] = f2bf(val);
            } else {
                const int f2 = f - 2 * E_;
                const int h = f2 >> 6, d = f2 & 63;
                Vt[(((size_t)(bb * H_ + h)) * D_ + d) * S_ + ss] = f2bf(val);
            }
        }
    }
}

// ---------------- output projection GEMM ----------------
__global__ __launch_bounds__(256)
void proj_gemm_kernel(const unsigned short* __restrict__ obf,
                      const unsigned short* __restrict__ wbf,
                      const float* __restrict__ bias,
                      float* __restrict__ out) {
    __shared__ unsigned short Wl[2 * WTILE];
    const int lane = threadIdx.x & 31;
    const int wave = threadIdx.x >> 5;
    const int half = lane >> 4;
    const int idx  = lane & 15;
    const int NG   = E_ / 64;                // 16
    const int ng   = blockIdx.x % NG;
    const int mb   = blockIdx.x / NG;        // 0..31
    const int m0   = mb * 128 + wave * 16;
    const int n0   = ng * 64;

    v8f acc[4] = {};
    gemm_core(obf, wbf, m0, n0, Wl, acc);

#pragma unroll
    for (int t = 0; t < 4; ++t) {
        const int f = n0 + t * 16 + idx;
        const float bv = bias[f];
#pragma unroll
        for (int r = 0; r < 8; ++r) {
            const int m = r + 8 * half;
            out[(size_t)(m0 + m) * E_ + f] = acc[t][r] + bv;
        }
    }
}

// ---------------- flash attention ----------------
// One wave (block of 32) per (b,h, 16-query tile); online softmax over
// 32-key blocks; causal mask computed arithmetically.
__global__ __launch_bounds__(32)
void attn_kernel(const unsigned short* __restrict__ Q,
                 const unsigned short* __restrict__ K,
                 const unsigned short* __restrict__ Vt,
                 unsigned short* __restrict__ O) {
    __shared__ unsigned short P[16 * 32];     // P tile, C-layout -> A-layout bounce

    const int lane = threadIdx.x & 31;
    const int half = lane >> 4;
    const int idx  = lane & 15;
    const int QT   = S_ / 16;                 // 128
    const int qt   = blockIdx.x % QT;
    const int bh   = blockIdx.x / QT;         // b*H + h
    const int q0   = qt * 16;

    const unsigned short* Qb = Q  + (size_t)bh * S_ * D_;
    const unsigned short* Kb = K  + (size_t)bh * S_ * D_;
    const unsigned short* Vb = Vt + (size_t)bh * D_ * S_;

    const unsigned short* qrow = Qb + (size_t)(q0 + idx) * D_ + 8 * half;
    const v16bf aq0 = ldfrag(qrow);
    const v16bf aq1 = ldfrag(qrow + 32);

    v8f o[4] = {};
    float mrow[8], lrow[8];
#pragma unroll
    for (int r = 0; r < 8; ++r) { mrow[r] = NEG_BIG; lrow[r] = 0.0f; }

    const int nkb = (q0 + 47) >> 5;           // ceil((q0+16)/32)
    for (int kb = 0; kb < nkb; ++kb) {
        const int kbase = kb * 32;

        // ---- S = Q K^T  (two 16-key halves) ----
        v8f slo = {}, shi = {};
        {
            const unsigned short* kr0 = Kb + (size_t)(kbase + idx) * D_ + 8 * half;
            slo = wmma_bf16(aq0, ldfrag(kr0),      slo);
            slo = wmma_bf16(aq1, ldfrag(kr0 + 32), slo);
            const unsigned short* kr1 = Kb + (size_t)(kbase + 16 + idx) * D_ + 8 * half;
            shi = wmma_bf16(aq0, ldfrag(kr1),      shi);
            shi = wmma_bf16(aq1, ldfrag(kr1 + 32), shi);
        }

        // ---- causal mask + block row-max ----
        float bm[8];
#pragma unroll
        for (int r = 0; r < 8; ++r) {
            const int qg = q0 + r + 8 * half;
            if (kbase + idx > qg)      slo[r] = NEG_BIG;
            if (kbase + 16 + idx > qg) shi[r] = NEG_BIG;
            bm[r] = fmaxf(slo[r], shi[r]);
        }
#pragma unroll
        for (int r = 0; r < 8; ++r)
            for (int sh = 1; sh < 16; sh <<= 1)
                bm[r] = fmaxf(bm[r], __shfl_xor(bm[r], sh, 32));

        // ---- online softmax update ----
        float alpha[8], rs[8];
#pragma unroll
        for (int r = 0; r < 8; ++r) {
            const float mn = fmaxf(mrow[r], bm[r]);
            alpha[r] = __expf(mrow[r] - mn);
            mrow[r]  = mn;
            const float pl = __expf(slo[r] - mn);
            const float ph = __expf(shi[r] - mn);
            slo[r] = pl; shi[r] = ph;
            rs[r] = pl + ph;
        }
#pragma unroll
        for (int r = 0; r < 8; ++r)
            for (int sh = 1; sh < 16; sh <<= 1)
                rs[r] += __shfl_xor(rs[r], sh, 32);
#pragma unroll
        for (int r = 0; r < 8; ++r) {
            lrow[r] = lrow[r] * alpha[r] + rs[r];
            o[0][r] *= alpha[r]; o[1][r] *= alpha[r];
            o[2][r] *= alpha[r]; o[3][r] *= alpha[r];
        }

        // ---- P: C-layout -> LDS -> A-layout ----
        __syncthreads();                      // prior P reads done
#pragma unroll
        for (int r = 0; r < 8; ++r) {
            const int m = r + 8 * half;
            P[m * 32 + idx]      = f2bf(slo[r]);
            P[m * 32 + 16 + idx] = f2bf(shi[r]);
        }
        __syncthreads();                      // stores visible
        Frag fp;
        const unsigned short* prow = &P[idx * 32 + 8 * half];
        fp.q[0] = *(const uint4*)(prow);
        fp.q[1] = *(const uint4*)(prow + 16);

        // ---- O += P V ----
#pragma unroll
        for (int t = 0; t < 4; ++t) {
            const unsigned short* vrow =
                Vb + (size_t)(t * 16 + idx) * S_ + kbase + 8 * half;
            o[t] = wmma_bf16(fp.v, ldfrag(vrow), o[t]);
        }
    }

    // ---- normalize + store O[B,S,E] ----
    const int bb = bh >> 4, h = bh & 15;
#pragma unroll
    for (int r = 0; r < 8; ++r) {
        const float inv = 1.0f / lrow[r];
        const int qg = q0 + r + 8 * half;
#pragma unroll
        for (int t = 0; t < 4; ++t) {
            const int d = t * 16 + idx;
            O[((size_t)bb * S_ + qg) * E_ + h * D_ + d] = f2bf(o[t][r] * inv);
        }
    }
}

// ---------------- host launch ----------------
extern "C" void kernel_launch(void* const* d_in, const int* in_sizes, int n_in,
                              void* d_out, int out_size, void* d_ws, size_t ws_size,
                              hipStream_t stream) {
    const float* x      = (const float*)d_in[0];   // [B,S,E]
    const float* W_attn = (const float*)d_in[1];   // [3E,E]
    const float* b_attn = (const float*)d_in[2];   // [3E]
    const float* W_proj = (const float*)d_in[3];   // [E,E]
    const float* b_proj = (const float*)d_in[4];   // [E]
    // d_in[5] = causal mask: recomputed arithmetically, unread.

    char* ws = (char*)d_ws;
    size_t off = 0;
    auto take = [&](size_t bytes) {
        void* p = ws + off;
        off += (bytes + 255) & ~(size_t)255;
        return p;
    };
    const size_t NX  = (size_t)B_ * S_ * E_;        // 4,194,304
    const size_t NWA = (size_t)3 * E_ * E_;         // 3,145,728
    const size_t NWP = (size_t)E_ * E_;             // 1,048,576
    const size_t NQKV = (size_t)B_ * H_ * S_ * D_;  // 4,194,304 each

    unsigned short* xbf = (unsigned short*)take(NX * 2);
    unsigned short* Wa  = (unsigned short*)take(NWA * 2);
    unsigned short* Wp  = (unsigned short*)take(NWP * 2);
    unsigned short* Qb  = (unsigned short*)take(NQKV * 2);
    unsigned short* Kb  = (unsigned short*)take(NQKV * 2);
    unsigned short* Vb  = (unsigned short*)take(NQKV * 2);
    unsigned short* Ob  = (unsigned short*)take(NX * 2);

    cvt_bf16_kernel<<<(int)(NX  / 256), 256, 0, stream>>>(x,      xbf, (int)NX);
    cvt_bf16_kernel<<<(int)(NWA / 256), 256, 0, stream>>>(W_attn, Wa,  (int)NWA);
    cvt_bf16_kernel<<<(int)(NWP / 256), 256, 0, stream>>>(W_proj, Wp,  (int)NWP);

    // QKV: 32 M-blocks (128 rows) * 48 N-groups (64 cols)
    qkv_gemm_kernel<<<32 * 48, 256, 0, stream>>>(xbf, Wa, b_attn, Qb, Kb, Vb);

    // attention: B*H * S/16 = 4096 waves, one per block
    attn_kernel<<<4096, 32, 0, stream>>>(Qb, Kb, Vb, Ob);

    // projection: 32 M-blocks * 16 N-groups
    proj_gemm_kernel<<<32 * 16, 256, 0, stream>>>(Ob, Wp, b_proj, (float*)d_out);
}
